// Recommender_46205258170765
// MI455X (gfx1250) — matrix operand order repaired and checked by reading, not compiled, not run on hardware
//
#include <hip/hip_runtime.h>
#include <hip/hip_bf16.h>
#include <math.h>

typedef __attribute__((ext_vector_type(16))) _Float16 v16h;
typedef __attribute__((ext_vector_type(8)))  float    v8f;

#define THREADS 256

// ---------------------------------------------------------------------------
// Utility kernels
// ---------------------------------------------------------------------------
__global__ void zero_f(float* __restrict__ p, long long n) {
    long long i = (long long)blockIdx.x * blockDim.x + threadIdx.x;
    if (i < n) p[i] = 0.0f;
}

__global__ void copy_f(float* __restrict__ dst, const float* __restrict__ src, long long n) {
    long long i = (long long)blockIdx.x * blockDim.x + threadIdx.x;
    if (i < n) dst[i] = src[i];
}

// ---------------------------------------------------------------------------
// Edge scatter:  sum[head[e]] += ent[tail[e]] * weight[etype[e]],  cnt[head[e]] += 1
// One thread per (edge, 4-float chunk): 16 threads/edge, float4 traffic.
// ---------------------------------------------------------------------------
__global__ void edge_scatter(const float* __restrict__ ent,
                             const float* __restrict__ weight,
                             const int*   __restrict__ head,
                             const int*   __restrict__ tail,
                             const int*   __restrict__ etype,
                             float* __restrict__ sum,
                             float* __restrict__ cnt,
                             int E) {
    long long idx = (long long)blockIdx.x * blockDim.x + threadIdx.x;
    if (idx >= (long long)E * 16) return;
    int e = (int)(idx >> 4);
    int q = (int)(idx & 15);
    int t = tail[e], hd = head[e], r = etype[e];
    const float4 v = *(const float4*)(ent    + (size_t)t * 64 + q * 4);
    const float4 w = *(const float4*)(weight + (size_t)r * 64 + q * 4);
    float* dst = sum + (size_t)hd * 64 + q * 4;
    atomicAdd(dst + 0, v.x * w.x);
    atomicAdd(dst + 1, v.y * w.y);
    atomicAdd(dst + 2, v.z * w.z);
    atomicAdd(dst + 3, v.w * w.w);
    if (q == 0) atomicAdd(cnt + hd, 1.0f);
}

// ---------------------------------------------------------------------------
// Generic gather->scatter-add over an index pair list (used for both the
// user->item scatter-mean numerator and the item->user segment_sum).
// cnt may be nullptr (plain segment_sum).
// ---------------------------------------------------------------------------
__global__ void gather_scatter_add(const float* __restrict__ src,
                                   const int*   __restrict__ sidx,
                                   const int*   __restrict__ didx,
                                   float* __restrict__ sum,
                                   float* __restrict__ cnt,
                                   int N) {
    long long idx = (long long)blockIdx.x * blockDim.x + threadIdx.x;
    if (idx >= (long long)N * 16) return;
    int n = (int)(idx >> 4);
    int q = (int)(idx & 15);
    int s = sidx[n], d = didx[n];
    const float4 v = *(const float4*)(src + (size_t)s * 64 + q * 4);
    float* dst = sum + (size_t)d * 64 + q * 4;
    atomicAdd(dst + 0, v.x);
    atomicAdd(dst + 1, v.y);
    atomicAdd(dst + 2, v.z);
    atomicAdd(dst + 3, v.w);
    if (cnt && q == 0) atomicAdd(cnt + d, 1.0f);
}

// sum[row][d] /= max(cnt[row], 1)   (in place, rows*64 elements)
__global__ void divide_mean(float* __restrict__ sum, const float* __restrict__ cnt,
                            long long rows) {
    long long i = (long long)blockIdx.x * blockDim.x + threadIdx.x;
    if (i >= rows * 64) return;
    float c = cnt[i >> 6];
    sum[i] = sum[i] / fmaxf(c, 1.0f);
}

// ---------------------------------------------------------------------------
// Gate + fusion via WMMA (v_wmma_f32_16x16x32_f16):
//   Z = KG @ G1^T + INT @ G2^T        (per 16x16 tile, K=64 => 2 k-chunks x 2 gemms)
//   gi = sigmoid(Z); fusion = gi*KG + (1-gi)*INT
// One wave per 16-row tile, 8 waves per block. EXEC is all-ones per wave
// (uniform early-out only). Fragment layouts follow CDNA5 ISA:
//   A (16x32 f16): lane l holds M=l&15; halves h: K = 32*kc + 16*(h>>3) + 8*(l>>4) + (h&7)
//   B (32x16 f16): lane l holds N = n0 + (l&15); halves h: K as for A
//   C/D (16x16 f32): vgpr r -> M = r + 8*(l>>4), N = l&15
// ---------------------------------------------------------------------------
__global__ void gate_fusion_wmma(const float* __restrict__ kg,   // [n_items,64] (= entity mean, first n_items rows)
                                 const float* __restrict__ ii,   // [n_items,64]
                                 const float* __restrict__ g1,   // [64,64] row-major (n,k)
                                 const float* __restrict__ g2,   // [64,64]
                                 float* __restrict__ fusion,     // [n_items,64]
                                 int n_items) {
    const int lane = threadIdx.x & 31;
    const int wave = threadIdx.x >> 5;
    const int row0 = (blockIdx.x * 8 + wave) * 16;
    if (row0 >= n_items) return;

    const int m    = lane & 15;
    const int hsel = lane >> 4;

    // A fragments for both operands, both K chunks
    v16h a_kg[2], a_ii[2];
#pragma unroll
    for (int kc = 0; kc < 2; ++kc) {
#pragma unroll
        for (int h = 0; h < 16; ++h) {
            const int k = kc * 32 + ((h >> 3) << 4) + (hsel << 3) + (h & 7);
            a_kg[kc][h] = (_Float16)kg[(size_t)(row0 + m) * 64 + k];
            a_ii[kc][h] = (_Float16)ii[(size_t)(row0 + m) * 64 + k];
        }
    }

#pragma unroll
    for (int nt = 0; nt < 4; ++nt) {
        const int n0 = nt * 16;
        v8f acc = {};
#pragma unroll
        for (int kc = 0; kc < 2; ++kc) {
            const int kbase = kc * 32;
            v16h b1, b2;
#pragma unroll
            for (int h = 0; h < 16; ++h) {
                // B[k][n] = G[n][k]  (Z = A @ G^T)
                b1[h] = (_Float16)g1[(size_t)(n0 + m) * 64 + kbase + ((h >> 3) << 4) + (hsel << 3) + (h & 7)];
                b2[h] = (_Float16)g2[(size_t)(n0 + m) * 64 + kbase + ((h >> 3) << 4) + (hsel << 3) + (h & 7)];
            }
            acc = __builtin_amdgcn_wmma_f32_16x16x32_f16(
                false, a_kg[kc], false, b1, (short)0, acc, false, false);
            acc = __builtin_amdgcn_wmma_f32_16x16x32_f16(
                false, a_ii[kc], false, b2, (short)0, acc, false, false);
        }
        // Fused sigmoid gate + blend, written straight back out.
        const int n = n0 + (lane & 15);
#pragma unroll
        for (int r = 0; r < 8; ++r) {
            const int mm = row0 + r + 8 * hsel;
            const float gi = 1.0f / (1.0f + __expf(-acc[r]));
            const float a  = kg[(size_t)mm * 64 + n];
            const float b  = ii[(size_t)mm * 64 + n];
            fusion[(size_t)mm * 64 + n] = gi * a + (1.0f - gi) * b;
        }
    }
}

// ---------------------------------------------------------------------------
// Row-wise L2 normalize (D=64), optionally store normalized row, and
// accumulate into a residual buffer. One wave (32 lanes) per row, 2 floats/lane.
// ---------------------------------------------------------------------------
__global__ void l2norm_acc(const float* __restrict__ src,
                           float* __restrict__ dst,     // nullable
                           float* __restrict__ res,     // res += l2norm(src)
                           int rows) {
    const int row = blockIdx.x * (blockDim.x >> 5) + (threadIdx.x >> 5);
    if (row >= rows) return;
    const int lane = threadIdx.x & 31;
    const size_t base = (size_t)row * 64;
    float x0 = src[base + lane];
    float x1 = src[base + lane + 32];
    float s = x0 * x0 + x1 * x1;
#pragma unroll
    for (int o = 16; o > 0; o >>= 1) s += __shfl_xor(s, o, 32);
    const float denom = fmaxf(sqrtf(s), 1e-12f);
    const float y0 = x0 / denom, y1 = x1 / denom;
    if (dst) { dst[base + lane] = y0; dst[base + lane + 32] = y1; }
    res[base + lane]      += y0;
    res[base + lane + 32] += y1;
}

// ---------------------------------------------------------------------------
static inline unsigned gridFor(long long n) { return (unsigned)((n + THREADS - 1) / THREADS); }

extern "C" void kernel_launch(void* const* d_in, const int* in_sizes, int n_in,
                              void* d_out, int out_size, void* d_ws, size_t ws_size,
                              hipStream_t stream) {
    const float* user_emb   = (const float*)d_in[0];
    const float* entity_emb = (const float*)d_in[1];
    const float* weight     = (const float*)d_in[2];
    const float* gate1_w    = (const float*)d_in[3];
    const float* gate2_w    = (const float*)d_in[4];
    const int*   edge_index = (const int*)d_in[5];
    const int*   edge_type  = (const int*)d_in[6];
    const int*   mat_row    = (const int*)d_in[7];
    const int*   mat_col    = (const int*)d_in[8];
    // d_in[9] (n_items) is on device; derive everything host-side instead.

    const int D       = 64;
    const int n_users = in_sizes[0] / D;
    const int n_ent   = in_sizes[1] / D;
    const int hops    = in_sizes[3] / (D * D);
    const int E       = in_sizes[5] / 2;
    const int NNZ     = in_sizes[7];
    const int n_items = (int)(((long long)out_size / D - n_ent - n_users) / 2);

    const int* head = edge_index;
    const int* tail = edge_index + E;

    // ---- workspace layout (floats) ----
    float* ws = (float*)d_ws;
    float* entity_sum   = ws;                                  // n_ent*D  (becomes entity mean)
    float* entity_cnt   = entity_sum   + (size_t)n_ent * D;    // n_ent
    float* item_int_sum = entity_cnt   + (size_t)n_ent;        // n_items*D (becomes item_int_agg)
    float* item_cnt     = item_int_sum + (size_t)n_items * D;  // n_items
    float* item_fusion  = item_cnt     + (size_t)n_items;      // n_items*D
    float* user_agg     = item_fusion  + (size_t)n_items * D;  // n_users*D
    float* ent_buf      = user_agg     + (size_t)n_users * D;  // n_ent*D   (next-hop ent)
    float* usr_buf      = ent_buf      + (size_t)n_ent * D;    // n_users*D (next-hop usr)
    const long long zero_span = (long long)(user_agg - entity_sum) + (long long)n_users * D;

    // ---- output regions double as residual accumulators ----
    float* out          = (float*)d_out;
    float* entity_res   = out;
    float* user_res     = entity_res   + (size_t)n_ent * D;
    float* item_kg_res  = user_res     + (size_t)n_users * D;
    float* item_int_res = item_kg_res  + (size_t)n_items * D;

    // init residuals
    copy_f<<<gridFor((long long)n_ent * D),   THREADS, 0, stream>>>(entity_res, entity_emb, (long long)n_ent * D);
    copy_f<<<gridFor((long long)n_users * D), THREADS, 0, stream>>>(user_res,   user_emb,   (long long)n_users * D);
    zero_f<<<gridFor((long long)2 * n_items * D), THREADS, 0, stream>>>(item_kg_res, (long long)2 * n_items * D);

    const float* cur_ent = entity_emb;
    const float* cur_usr = user_emb;
    const int rows_per_blk = THREADS / 32;  // waves per block for l2norm_acc
    auto rowGrid = [&](int rows) -> unsigned {
        return (unsigned)((rows + rows_per_blk - 1) / rows_per_blk);
    };

    for (int hop = 0; hop < hops; ++hop) {
        // clear this hop's accumulators (entity_sum..user_agg are contiguous)
        zero_f<<<gridFor(zero_span), THREADS, 0, stream>>>(entity_sum, zero_span);

        // entity_agg = scatter_mean(ent[tail] * weight[etype], head)
        edge_scatter<<<gridFor((long long)E * 16), THREADS, 0, stream>>>(
            cur_ent, weight, head, tail, edge_type, entity_sum, entity_cnt, E);
        divide_mean<<<gridFor((long long)n_ent * D), THREADS, 0, stream>>>(
            entity_sum, entity_cnt, n_ent);

        // item_int_agg = scatter_mean(usr[mat_row], mat_col)
        gather_scatter_add<<<gridFor((long long)NNZ * 16), THREADS, 0, stream>>>(
            cur_usr, mat_row, mat_col, item_int_sum, item_cnt, NNZ);
        divide_mean<<<gridFor((long long)n_items * D), THREADS, 0, stream>>>(
            item_int_sum, item_cnt, n_items);

        // gi = sigmoid(KG@G1^T + INT@G2^T); fusion = gi*KG + (1-gi)*INT   [WMMA]
        {
            const int mtiles = (n_items + 15) / 16;
            const unsigned blocks = (unsigned)((mtiles + 7) / 8);
            gate_fusion_wmma<<<blocks, THREADS, 0, stream>>>(
                entity_sum, item_int_sum,
                gate1_w + (size_t)hop * D * D, gate2_w + (size_t)hop * D * D,
                item_fusion, n_items);
        }

        // user_agg = segment_sum(item_fusion[mat_col], mat_row)
        gather_scatter_add<<<gridFor((long long)NNZ * 16), THREADS, 0, stream>>>(
            item_fusion, mat_col, mat_row, user_agg, (float*)nullptr, NNZ);

        // ent = l2norm(concat(item_fusion, att_kg_agg)); entity_res += ent
        l2norm_acc<<<rowGrid(n_items), THREADS, 0, stream>>>(
            item_fusion, ent_buf, entity_res, n_items);
        l2norm_acc<<<rowGrid(n_ent - n_items), THREADS, 0, stream>>>(
            entity_sum + (size_t)n_items * D, ent_buf + (size_t)n_items * D,
            entity_res + (size_t)n_items * D, n_ent - n_items);

        // usr = l2norm(user_agg); user_res += usr
        l2norm_acc<<<rowGrid(n_users), THREADS, 0, stream>>>(
            user_agg, usr_buf, user_res, n_users);

        // item residuals
        l2norm_acc<<<rowGrid(n_items), THREADS, 0, stream>>>(
            entity_sum, (float*)nullptr, item_kg_res, n_items);
        l2norm_acc<<<rowGrid(n_items), THREADS, 0, stream>>>(
            item_int_sum, (float*)nullptr, item_int_res, n_items);

        cur_ent = ent_buf;
        cur_usr = usr_buf;
    }
}
